// Projection_67293547594093
// MI455X (gfx1250) — matrix operand order repaired and checked by reading, not compiled, not run on hardware
//
#include <hip/hip_runtime.h>
#include <hip/hip_bf16.h>

// Problem constants (from reference)
#define D_   4096
#define H_   32
#define Q_   96
#define KV_  1152
#define BS_  4
#define L_   4096     // GMAX*GMAX
#define HD_  128      // D_/H_
#define PEW  70       // pos-embed cache side

// GEMM block tile: 128x128, BK=64, 8 wave32s each computing a 16x128 strip
#define BM_  128
#define BN_  128
#define BK_  64
#define LDP_ 72       // padded LDS row stride (bf16 elems): 144B, 16B-aligned, conflict-free

typedef __attribute__((ext_vector_type(16))) __bf16 v16bf;
typedef __attribute__((ext_vector_type(8)))  float  v8f;
typedef int v4i_ __attribute__((vector_size(16)));   // matches async builtin's pointer element type

union Frag { uint4 q[2]; v16bf v; };
union Pack8 { uint4 q; unsigned short s[8]; };

__device__ __forceinline__ unsigned short f2bf(float f) {
    unsigned int u = __float_as_uint(f);
    unsigned int r = u + 0x7FFFu + ((u >> 16) & 1u);   // round-to-nearest-even
    return (unsigned short)(r >> 16);
}
__device__ __forceinline__ float bf2f(unsigned short h) {
    return __uint_as_float(((unsigned int)h) << 16);
}

// ---- CDNA5 async global->LDS copy (ASYNCcnt-tracked path), 16B per lane
__device__ __forceinline__ void cp_async16(const unsigned short* gsrc, unsigned short* ldst) {
#if __has_builtin(__builtin_amdgcn_global_load_async_to_lds_b128)
    __builtin_amdgcn_global_load_async_to_lds_b128(
        (__attribute__((address_space(1))) v4i_*)gsrc,
        (__attribute__((address_space(3))) v4i_*)ldst, 0, 0);
#else
    *(uint4*)ldst = *(const uint4*)gsrc;
#endif
}
__device__ __forceinline__ void wait_async0() {
#if __has_builtin(__builtin_amdgcn_s_wait_asynccnt)
    __builtin_amdgcn_s_wait_asynccnt(0);
#else
    asm volatile("s_wait_asynccnt 0x0" ::: "memory");
#endif
}

// ---------------------------------------------------------------- converts
__global__ void k_cvt(const float* __restrict__ in, unsigned short* __restrict__ out, long n) {
    long i = blockIdx.x * (long)blockDim.x + threadIdx.x;
    long stride = (long)gridDim.x * blockDim.x;
    for (; i < n; i += stride) out[i] = f2bf(in[i]);
}

// proj is used as  x @ proj  (NOT proj.T) -> store transposed so GEMM reads rows along K
__global__ void k_cvt_tr(const float* __restrict__ in, unsigned short* __restrict__ out) {
    long n = (long)D_ * D_;
    long i = blockIdx.x * (long)blockDim.x + threadIdx.x;
    long stride = (long)gridDim.x * blockDim.x;
    for (; i < n; i += stride) {
        int k = (int)(i >> 12);          // row of proj
        int c = (int)(i & (D_ - 1));     // col of proj
        out[(long)c * D_ + k] = f2bf(in[i]);
    }
}

// ---------------------------------------------------------------- LayerNorm (fp32 in -> bf16 out), one block per row, N = 4096
__global__ void k_ln(const float* __restrict__ src, const float* __restrict__ g,
                     const float* __restrict__ bt, unsigned short* __restrict__ dst) {
    __shared__ float s1[256], s2[256];
    int row = blockIdx.x, tid = threadIdx.x;
    const float* p = src + (long)row * D_;
    float v[16], a = 0.f, b = 0.f;
#pragma unroll
    for (int i = 0; i < 16; i++) { v[i] = p[tid + i * 256]; a += v[i]; b += v[i] * v[i]; }
    s1[tid] = a; s2[tid] = b; __syncthreads();
    for (int o = 128; o > 0; o >>= 1) {
        if (tid < o) { s1[tid] += s1[tid + o]; s2[tid] += s2[tid + o]; }
        __syncthreads();
    }
    float mean = s1[0] * (1.f / D_);
    float var  = s2[0] * (1.f / D_) - mean * mean;
    float r    = rsqrtf(var + 1e-6f);
    unsigned short* o_ = dst + (long)row * D_;
#pragma unroll
    for (int i = 0; i < 16; i++) {
        int c = tid + i * 256;
        o_[c] = f2bf((v[i] - mean) * r * g[c] + bt[c]);
    }
}

// ---------------------------------------------------------------- LN of kv-proj (bf16 in) + pos-embed gather -> kin/vin bf16
__global__ void k_lnpe(const unsigned short* __restrict__ xkv, const int* __restrict__ ts,
                       const float* __restrict__ pe, const float* __restrict__ g,
                       const float* __restrict__ bt,
                       unsigned short* __restrict__ kin, unsigned short* __restrict__ vin) {
    __shared__ float s1[256], s2[256];
    int row = blockIdx.x, tid = threadIdx.x;
    int b = row >> 12, p = row & (L_ - 1);
    int th = ts[2 * b], tw = ts[2 * b + 1];
    int plen = th * tw;
    int wi = tw > 1 ? tw : 1;
    int hidx = p / wi; if (hidx > PEW - 1) hidx = PEW - 1;
    int widx = p % wi;
    bool valid = p < plen;
    const float* per = pe + ((long)(hidx * PEW + widx)) * D_;
    const unsigned short* src = xkv + (long)row * D_;
    float v[16], a = 0.f, bb = 0.f;
#pragma unroll
    for (int i = 0; i < 16; i++) { v[i] = bf2f(src[tid + i * 256]); a += v[i]; bb += v[i] * v[i]; }
    s1[tid] = a; s2[tid] = bb; __syncthreads();
    for (int o = 128; o > 0; o >>= 1) {
        if (tid < o) { s1[tid] += s1[tid + o]; s2[tid] += s2[tid + o]; }
        __syncthreads();
    }
    float mean = s1[0] * (1.f / D_);
    float var  = s2[0] * (1.f / D_) - mean * mean;
    float r    = rsqrtf(var + 1e-6f);
#pragma unroll
    for (int i = 0; i < 16; i++) {
        int c = tid + i * 256;
        float y = (v[i] - mean) * r * g[c] + bt[c];
        vin[(long)row * D_ + c] = f2bf(y);
        float pv = valid ? per[c] : 0.f;
        kin[(long)row * D_ + c] = f2bf(y + pv);
    }
}

// ---------------------------------------------------------------- bf16 WMMA GEMM, 128x128 tile, async double-buffered LDS staging
// out[M,N] = (A[M,K] @ W[N,K]^T + bias) * alpha     (M-edge via clamp/guard, N,K exact multiples)
__device__ __forceinline__ void stage_tile(const unsigned short* __restrict__ A,
                                           const unsigned short* __restrict__ W,
                                           int M, int K, int mblk, int nblk, int kb,
                                           unsigned short* sA, unsigned short* sW, int tid) {
    // A and W tiles: 128 rows x 64 K each = 1024 x 16B transfers apiece (4+4 per thread)
#pragma unroll
    for (int i = 0; i < 4; i++) {
        int t = tid + i * 256;           // 0..1023
        int m = t >> 3, k8 = (t & 7) * 8;
        int mrow = mblk + m; if (mrow > M - 1) mrow = M - 1;   // clamp ragged M (stores are guarded)
        cp_async16(A + (long)mrow * K + kb + k8, sA + m * LDP_ + k8);
        cp_async16(W + (long)(nblk + m) * K + kb + k8, sW + m * LDP_ + k8);
    }
}

__global__ void __launch_bounds__(256)
k_gemm(const unsigned short* __restrict__ A, const unsigned short* __restrict__ W,
       const float* __restrict__ bias, void* __restrict__ out,
       int M, int N, int K, float alpha, int out_bf16) {
    __shared__ __align__(16) unsigned short lA[2][BM_ * LDP_];
    __shared__ __align__(16) unsigned short lW[2][BN_ * LDP_];
    int tid  = threadIdx.x;
    int lane = tid & 31, wv = tid >> 5;
    int l15 = lane & 15, hi = lane >> 4;
    int mblk = blockIdx.y * BM_;
    int nblk = blockIdx.x * BN_;

    v8f acc[8];
#pragma unroll
    for (int t = 0; t < 8; t++)
#pragma unroll
        for (int i = 0; i < 8; i++) acc[t][i] = 0.f;

    // prologue: stage k-block 0 into buffer 0
    stage_tile(A, W, M, K, mblk, nblk, 0, lA[0], lW[0], tid);
    wait_async0();
    __syncthreads();

    int nk = K / BK_;
    for (int ib = 0; ib < nk; ib++) {
        int buf = ib & 1;
        if (ib + 1 < nk)   // async-stage next k-block into the other buffer while computing
            stage_tile(A, W, M, K, mblk, nblk, (ib + 1) * BK_, lA[buf ^ 1], lW[buf ^ 1], tid);

#pragma unroll
        for (int kk = 0; kk < BK_; kk += 32) {
            Frag a;
            const unsigned short* ar = &lA[buf][(wv * 16 + l15) * LDP_ + kk];
            a.q[0] = *(const uint4*)(ar + hi * 8);
            a.q[1] = *(const uint4*)(ar + 16 + hi * 8);
            // 8 n-tiles in two batches of 4: one dscnt wait covers 4 back-to-back WMMAs
#pragma unroll
            for (int g = 0; g < 2; g++) {
                Frag bf[4];
#pragma unroll
                for (int t = 0; t < 4; t++) {
                    const unsigned short* wr =
                        &lW[buf][((g * 4 + t) * 16 + l15) * LDP_ + kk + hi * 8];
                    bf[t].q[0] = *(const uint4*)(wr);
                    bf[t].q[1] = *(const uint4*)(wr + 16);
                }
#pragma unroll
                for (int t = 0; t < 4; t++)
                    acc[g * 4 + t] = __builtin_amdgcn_wmma_f32_16x16x32_bf16(
                        false, a.v, false, bf[t].v, (short)0, acc[g * 4 + t], false, false);
            }
        }
        if (ib + 1 < nk) wait_async0();
        __syncthreads();
    }

#pragma unroll
    for (int t = 0; t < 8; t++) {
        int col = nblk + t * 16 + l15;
        float bv = bias ? bias[col] : 0.f;
#pragma unroll
        for (int r = 0; r < 8; r++) {
            int row = mblk + wv * 16 + r + hi * 8;
            if (row < M) {
                float val = (acc[t][r] + bv) * alpha;
                if (out_bf16) ((unsigned short*)out)[(long)row * N + col] = f2bf(val);
                else          ((float*)out)[(long)row * N + col] = val;
            }
        }
    }
}

// ---------------------------------------------------------------- flash-style attention, one block per (b,h), 6 wave32s (96 rows)
// Qm [Q_,D_] bf16 already scaled by hd^-0.5 and biased. Km/Vm [BS_,L_,D_] bf16.
__global__ void __launch_bounds__(192)
k_attn(const unsigned short* __restrict__ Qm, const unsigned short* __restrict__ Km,
       const unsigned short* __restrict__ Vm, const int* __restrict__ ts,
       unsigned short* __restrict__ O) {
    __shared__ __align__(16) unsigned short vT[128 * LDP_];     // V chunk transposed [hd][key]
    __shared__ __align__(16) unsigned short pl[6 * 16 * LDP_];  // per-wave P tiles (padded rows)

    int b = blockIdx.x >> 5;        // H_ = 32
    int h = blockIdx.x & 31;
    int tid  = threadIdx.x;
    int lane = tid & 31, wv = tid >> 5;
    int l15 = lane & 15, hi = lane >> 4;
    int plen = ts[2 * b] * ts[2 * b + 1];

    // Q A-frags for K-dim = 128 (4 k-steps of 32), held in registers
    Frag qa[4];
    {
        const unsigned short* qr = Qm + (long)(wv * 16 + l15) * D_ + h * HD_;
#pragma unroll
        for (int k4 = 0; k4 < 4; k4++) {
            qa[k4].q[0] = *(const uint4*)(qr + k4 * 32 + hi * 8);
            qa[k4].q[1] = *(const uint4*)(qr + k4 * 32 + 16 + hi * 8);
        }
    }

    v8f o[8];
#pragma unroll
    for (int t = 0; t < 8; t++)
#pragma unroll
        for (int i = 0; i < 8; i++) o[t][i] = 0.f;
    float mrow[8], lrow[8];
#pragma unroll
    for (int r = 0; r < 8; r++) { mrow[r] = -1e30f; lrow[r] = 0.f; }

    const unsigned short* Kb = Km + (long)b * L_ * D_ + h * HD_;
    const unsigned short* Vb = Vm + (long)b * L_ * D_ + h * HD_;
    unsigned short* pw = pl + wv * (16 * LDP_);

    for (int c0 = 0; c0 < L_; c0 += 64) {
        __syncthreads();                       // vT from previous chunk fully consumed
        // stage V chunk transposed into LDS (coalesced 16B reads)
        for (int t = tid; t < 1024; t += 192) {
            int key = t & 63, grp = t >> 6;    // grp: 16 groups of 8 hd-cols
            Pack8 tmp;
            tmp.q = *(const uint4*)(Vb + (long)(c0 + key) * D_ + grp * 8);
#pragma unroll
            for (int j = 0; j < 8; j++) vT[(grp * 8 + j) * LDP_ + key] = tmp.s[j];
        }
        __syncthreads();

        // scores S = qK^T for 16 rows x 64 keys (B-frags read straight from global: contiguous along hd)
        v8f st[4];
#pragma unroll
        for (int t = 0; t < 4; t++) {
            v8f a;
#pragma unroll
            for (int i = 0; i < 8; i++) a[i] = 0.f;
            int key = c0 + t * 16 + l15;
            const unsigned short* kr = Kb + (long)key * D_;
            __builtin_prefetch(kr + 64 * D_, 0, 1);    // next chunk's K rows
#pragma unroll
            for (int k4 = 0; k4 < 4; k4++) {
                Frag bf_;
                bf_.q[0] = *(const uint4*)(kr + k4 * 32 + hi * 8);
                bf_.q[1] = *(const uint4*)(kr + k4 * 32 + 16 + hi * 8);
                a = __builtin_amdgcn_wmma_f32_16x16x32_bf16(
                    false, qa[k4].v, false, bf_.v, (short)0, a, false, false);
            }
            st[t] = a;
        }

        // online softmax: mask, row max/sum reductions across the 16-lane column group
#pragma unroll
        for (int r = 0; r < 8; r++) {
            float sv[4], mx = -1e30f;
#pragma unroll
            for (int t = 0; t < 4; t++) {
                int key = c0 + t * 16 + l15;
                float s = (key < plen) ? st[t][r] : -1e30f;
                sv[t] = s; mx = fmaxf(mx, s);
            }
            for (int off = 1; off < 16; off <<= 1) mx = fmaxf(mx, __shfl_xor(mx, off, 32));
            float mnew = fmaxf(mrow[r], mx);
            float cs = __expf(mrow[r] - mnew);
            mrow[r] = mnew;
            lrow[r] *= cs;
#pragma unroll
            for (int t = 0; t < 8; t++) o[t][r] *= cs;
            float ssum = 0.f;
#pragma unroll
            for (int t = 0; t < 4; t++) {
                int key = c0 + t * 16 + l15;
                float pv = (key < plen) ? __expf(sv[t] - mnew) : 0.f;
                ssum += pv;
                pw[(r + hi * 8) * LDP_ + t * 16 + l15] = f2bf(pv);   // C-layout -> row-major P in LDS
            }
            for (int off = 1; off < 16; off <<= 1) ssum += __shfl_xor(ssum, off, 32);
            lrow[r] += ssum;
        }
        __syncthreads();                       // P visible before A-frag reads

        // O += P @ Vchunk : A-frags from P tile in LDS, B-frags from transposed V in LDS
#pragma unroll
        for (int k2 = 0; k2 < 2; k2++) {
            Frag pa;
            const unsigned short* pr = pw + l15 * LDP_ + k2 * 32;
            pa.q[0] = *(const uint4*)(pr + hi * 8);
            pa.q[1] = *(const uint4*)(pr + 16 + hi * 8);
#pragma unroll
            for (int t = 0; t < 8; t++) {
                const unsigned short* vr = vT + (t * 16 + l15) * LDP_ + k2 * 32 + hi * 8;
                Frag bf_;
                bf_.q[0] = *(const uint4*)(vr);
                bf_.q[1] = *(const uint4*)(vr + 16);
                o[t] = __builtin_amdgcn_wmma_f32_16x16x32_bf16(
                    false, pa.v, false, bf_.v, (short)0, o[t], false, false);
            }
        }
    }

    // epilogue: divide by row sums, write bf16 [b, q, h*128+d]
#pragma unroll
    for (int t = 0; t < 8; t++) {
        int col = h * HD_ + t * 16 + l15;
#pragma unroll
        for (int r = 0; r < 8; r++) {
            int row = wv * 16 + r + hi * 8;
            float val = o[t][r] / lrow[r];
            O[((long)(b * Q_ + row)) * D_ + col] = f2bf(val);
        }
    }
}

// ---------------------------------------------------------------- host
extern "C" void kernel_launch(void* const* d_in, const int* in_sizes, int n_in,
                              void* d_out, int out_size, void* d_ws, size_t ws_size,
                              hipStream_t stream) {
    (void)in_sizes; (void)n_in; (void)out_size; (void)ws_size;
    const float* x         = (const float*)d_in[0];
    const int*   ts        = (const int*)d_in[1];
    const float* query     = (const float*)d_in[2];
    const float* kv_w      = (const float*)d_in[3];
    const float* ln_q_w    = (const float*)d_in[4];
    const float* ln_q_b    = (const float*)d_in[5];
    const float* ln_kv_w   = (const float*)d_in[6];
    const float* ln_kv_b   = (const float*)d_in[7];
    const float* ln_post_w = (const float*)d_in[8];
    const float* ln_post_b = (const float*)d_in[9];
    const float* in_proj_w = (const float*)d_in[10];
    const float* in_proj_b = (const float*)d_in[11];
    const float* out_w     = (const float*)d_in[12];
    const float* out_b     = (const float*)d_in[13];
    const float* proj      = (const float*)d_in[14];
    const float* pos_embed = (const float*)d_in[15];

    char* wsb = (char*)d_ws;
    size_t off = 0;
    auto alloc = [&](size_t bytes) -> char* {
        char* p = wsb + off;
        off += (bytes + 255) & ~(size_t)255;
        return p;
    };
    unsigned short* X16    = (unsigned short*)alloc((size_t)BS_ * L_ * KV_ * 2);
    unsigned short* KVW16  = (unsigned short*)alloc((size_t)D_ * KV_ * 2);
    unsigned short* INP16  = (unsigned short*)alloc((size_t)3 * D_ * D_ * 2);   // wq|wk|wv
    unsigned short* OUTW16 = (unsigned short*)alloc((size_t)D_ * D_ * 2);
    unsigned short* PROJ16 = (unsigned short*)alloc((size_t)D_ * D_ * 2);       // transposed
    unsigned short* QN16   = (unsigned short*)alloc((size_t)Q_ * D_ * 2);
    unsigned short* Q16    = (unsigned short*)alloc((size_t)Q_ * D_ * 2);
    unsigned short* XKVP16 = (unsigned short*)alloc((size_t)BS_ * L_ * D_ * 2);
    unsigned short* KIN16  = (unsigned short*)alloc((size_t)BS_ * L_ * D_ * 2);
    unsigned short* VIN16  = (unsigned short*)alloc((size_t)BS_ * L_ * D_ * 2);
    unsigned short* K16    = (unsigned short*)alloc((size_t)BS_ * L_ * D_ * 2);
    unsigned short* V16    = (unsigned short*)alloc((size_t)BS_ * L_ * D_ * 2);
    unsigned short* AO16   = (unsigned short*)alloc((size_t)BS_ * Q_ * D_ * 2);
    float*          O2     = (float*)alloc((size_t)BS_ * Q_ * D_ * 4);
    unsigned short* O2LN16 = (unsigned short*)alloc((size_t)BS_ * Q_ * D_ * 2);

    const unsigned short* WQ16 = INP16;
    const unsigned short* WK16 = INP16 + (size_t)D_ * D_;
    const unsigned short* WV16 = INP16 + (size_t)2 * D_ * D_;
    const float scale = 0.08838834764831845f;   // 128^-0.5

    // fp32 -> bf16 conversions
    k_cvt<<<2048, 256, 0, stream>>>(x, X16, (long)BS_ * L_ * KV_);
    k_cvt<<<2048, 256, 0, stream>>>(kv_w, KVW16, (long)D_ * KV_);
    k_cvt<<<2048, 256, 0, stream>>>(in_proj_w, INP16, (long)3 * D_ * D_);
    k_cvt<<<2048, 256, 0, stream>>>(out_w, OUTW16, (long)D_ * D_);
    k_cvt_tr<<<2048, 256, 0, stream>>>(proj, PROJ16);

    // qn = LN(query);  q = (qn @ wq.T + bq) * scale  (bf16)
    k_ln<<<Q_, 256, 0, stream>>>(query, ln_q_w, ln_q_b, QN16);
    k_gemm<<<dim3(D_ / BN_, (Q_ + BM_ - 1) / BM_), 256, 0, stream>>>(
        QN16, WQ16, in_proj_b, Q16, Q_, D_, D_, scale, 1);

    // xkv_pre = x @ kv_w.T
    k_gemm<<<dim3(D_ / BN_, (BS_ * L_) / BM_), 256, 0, stream>>>(
        X16, KVW16, nullptr, XKVP16, BS_ * L_, D_, KV_, 1.0f, 1);

    // LN + pos-embed gather -> kin (xkv+pe), vin (xkv)
    k_lnpe<<<BS_ * L_, 256, 0, stream>>>(XKVP16, ts, pos_embed, ln_kv_w, ln_kv_b, KIN16, VIN16);

    // K = kin @ wk.T + bk ;  V = vin @ wv.T + bv
    k_gemm<<<dim3(D_ / BN_, (BS_ * L_) / BM_), 256, 0, stream>>>(
        KIN16, WK16, in_proj_b + D_, K16, BS_ * L_, D_, D_, 1.0f, 1);
    k_gemm<<<dim3(D_ / BN_, (BS_ * L_) / BM_), 256, 0, stream>>>(
        VIN16, WV16, in_proj_b + 2 * D_, V16, BS_ * L_, D_, D_, 1.0f, 1);

    // attention -> AO16 [BS, Q, D] bf16
    k_attn<<<BS_ * H_, 192, 0, stream>>>(Q16, K16, V16, ts, AO16);

    // out = AO @ out_w.T + out_b (fp32) ; LN ; @ proj (fp32 result)
    k_gemm<<<dim3(D_ / BN_, (BS_ * Q_) / BM_), 256, 0, stream>>>(
        AO16, OUTW16, out_b, O2, BS_ * Q_, D_, D_, 1.0f, 0);
    k_ln<<<BS_ * Q_, 256, 0, stream>>>(O2, ln_post_w, ln_post_b, O2LN16);
    k_gemm<<<dim3(D_ / BN_, (BS_ * Q_) / BM_), 256, 0, stream>>>(
        O2LN16, PROJ16, nullptr, d_out, BS_ * Q_, D_, D_, 1.0f, 0);
}
